// Model_13039520710807
// MI455X (gfx1250) — compile-verified, matrix operand order checked by reading, not run on hardware
//
#include <hip/hip_runtime.h>

typedef __attribute__((ext_vector_type(16))) __bf16 v16bf;
typedef __attribute__((ext_vector_type(8)))  float  v8f;

#define ROWS  32768   // B*N
#define SEQ   2048
#define BATCH 16
#define NH    8
#define DH    64
#define DM    512
#define NF    266
#define FP    320     // features padded to multiple of 64
#define FF    2048
#define BH    128     // BATCH*NH
#define NCLS  10

#define RATIO     0.06131393394849658f   // 266^-0.5
#define DATA_NORM 0.35355339059327378f   // 64^-0.25
#define DIAG_SCL  0.0625f                // 0.5 * 64^-0.5
#define SQRT_DM   22.627416997969522f

struct U128 { unsigned x, y, z, w; };

__device__ inline unsigned encMax(float f) {
  unsigned u = __float_as_uint(f);
  return (u & 0x80000000u) ? ~u : (u | 0x80000000u);
}
__device__ inline float decMax(unsigned u) {
  return (u & 0x80000000u) ? __uint_as_float(u ^ 0x80000000u)
                           : __uint_as_float(~u);
}

// Async global->LDS 16-byte copy (CDNA5, ASYNCcnt-tracked). The low 32 bits of
// a generic pointer into LDS are the LDS byte offset (ISA aperture rule).
__device__ inline void async_b128(const __bf16* g, __bf16* l) {
  unsigned lo = (unsigned)(size_t)l;
  unsigned long long ga = (unsigned long long)(size_t)g;
  asm volatile("global_load_async_to_lds_b128 %0, %1, off" :: "v"(lo), "v"(ga) : "memory");
}
__device__ inline void wait_async() {
  asm volatile("s_wait_asynccnt 0x0" ::: "memory");
}

// ---------------------------------------------------------------------------
// Generic bf16 WMMA GEMM:  C[M,Nn] = A[M,K] * Bt[Nn,K]^T   (both row-major, K
// contiguous). Block tile 128 x (TN*16), BK=32, 8 waves, double-buffered LDS
// filled with async global->LDS loads overlapped with WMMA compute.
// ---------------------------------------------------------------------------
struct GemmP {
  const __bf16* A; const __bf16* Bt;
  long batchA, batchB, batchOut, batchDiag, batchAmax;
  int  lda, ldb, ldo, M, Nn, K, maxcols;
  float* outf; __bf16* outb;
  const float* bias; const float* diag;  // diag doubles as d_inv for MODE 8
  unsigned* amax;
};

// MODE: 0=f32(+bias)  1=f32 bias+residual(in-place)  2=bf16 bias+GELU
//       3=row atomic max  4=global atomic max
//       5=qp epilogue (row max)  6=kp^T epilogue (global max, transposed store)
//       7=transposed bf16 store (ctx^T)  8=attn: scale by d_inv, scatter bf16
template<int MODE, int TN>
__global__ __launch_bounds__(256)
void perf_gemm(GemmP P) {
  constexpr int BN = TN * 16;
  constexpr int ASTR = 128 * 40;  // one A buffer, elements
  constexpr int BSTR = BN * 40;   // one B buffer, elements
  __shared__ __bf16 As[2 * ASTR];
  __shared__ __bf16 Bs[2 * BSTR];
  const int m0 = blockIdx.y * 128, n0 = blockIdx.x * BN;
  const long z = blockIdx.z;
  const __bf16* Ab = P.A + z * P.batchA;
  const __bf16* Bb = P.Bt + z * P.batchB;
  const int tid = threadIdx.x;
  const int lane = tid & 31, wv = tid >> 5;
  const int nl = lane & 15, kh = lane >> 4;

  // Guard-free staging: clamp out-of-range rows/cols to a valid address; the
  // resulting garbage accumulator rows/cols are discarded by epilogue guards.
  const int r = tid >> 1, half = tid & 1;
  const int gaRow = (m0 + r < P.M) ? (m0 + r) : (P.M - 1);
  const long aOff = (long)gaRow * P.lda + half * 16;
  const bool bAct = (tid < 2 * BN);           // uniform per wave
  const int gbRow = bAct ? ((n0 + r < P.Nn) ? (n0 + r) : (P.Nn - 1)) : 0;
  const long bOff = (long)gbRow * P.ldb + half * 16;

  auto issue = [&](int s, int buf) {
    const int k0 = s * 32;
    const __bf16* ga = Ab + aOff + k0;
    __bf16* la = As + buf * ASTR + r * 40 + half * 16;
    async_b128(ga, la);
    async_b128(ga + 8, la + 8);
    if (bAct) {
      const __bf16* gb = Bb + bOff + k0;
      __bf16* lb = Bs + buf * BSTR + r * 40 + half * 16;
      async_b128(gb, lb);
      async_b128(gb + 8, lb + 8);
    }
  };

  v8f acc[TN] = {};
  const int steps = P.K >> 5;

  issue(0, 0);
  wait_async();
  __syncthreads();

  for (int s = 0; s < steps; s++) {
    const int cur = s & 1;
    if (s + 1 < steps) issue(s + 1, cur ^ 1);  // prefetch next tile (async)

    union Frag { U128 u[2]; v16bf v; } fa, fb;
    const __bf16* ar = As + cur * ASTR + (wv * 16 + nl) * 40 + kh * 8;
    fa.u[0] = *(const U128*)(ar);
    fa.u[1] = *(const U128*)(ar + 16);
#pragma unroll
    for (int t = 0; t < TN; t++) {
      const __bf16* br = Bs + cur * BSTR + (t * 16 + nl) * 40 + kh * 8;
      fb.u[0] = *(const U128*)(br);
      fb.u[1] = *(const U128*)(br + 16);
      acc[t] = __builtin_amdgcn_wmma_f32_16x16x32_bf16(
          false, fa.v, false, fb.v, (short)0, acc[t], false, false);
    }
    wait_async();
    __syncthreads();
  }

  // ------------------------- epilogues -------------------------
  if (MODE == 0 || MODE == 1 || MODE == 2 || MODE >= 5) {
#pragma unroll
    for (int t = 0; t < TN; t++) {
      int col = n0 + t * 16 + nl;
#pragma unroll
      for (int j = 0; j < 8; j++) {
        int row = m0 + wv * 16 + kh * 8 + j;
        if (row >= P.M || col >= P.Nn) continue;
        float v = acc[t][j];
        if (MODE == 0) {
          float o = v + (P.bias ? P.bias[col] : 0.0f);
          P.outf[z * P.batchOut + (long)row * P.ldo + col] = o;
        } else if (MODE == 1) {
          float* p = P.outf + (long)row * P.ldo + col;
          *p = *p + v + P.bias[col];
        } else if (MODE == 2) {
          float xg = v + P.bias[col];
          float gl = 0.5f * xg * (1.0f + erff(xg * 0.7071067811865475f));
          P.outb[(long)row * P.ldo + col] = (__bf16)gl;
        } else if (MODE == 5) {
          float mx = decMax(P.amax[z * P.batchAmax + row]);
          float dg = P.diag[z * P.batchDiag + row];
          float rr = (col < P.maxcols) ? RATIO * (expf(v - dg - mx) + 1e-4f) : 0.0f;
          P.outb[z * P.batchOut + (long)row * P.ldo + col] = (__bf16)rr;
        } else if (MODE == 6) {
          float mx = decMax(P.amax[0]);
          float dg = P.diag[z * P.batchDiag + row];
          float rr = (col < P.maxcols) ? RATIO * (expf(v - dg - mx) + 1e-4f) : 0.0f;
          P.outb[z * P.batchOut + (long)col * P.ldo + row] = (__bf16)rr;
        } else if (MODE == 7) {
          P.outb[z * P.batchOut + (long)col * P.ldo + row] = (__bf16)v;
        } else if (MODE == 8) {
          float di = P.diag[z * P.batchDiag + row];
          long b = z >> 3, h = z & 7;
          P.outb[((b * SEQ + row) * DM) + h * DH + col] = (__bf16)(v * di);
        }
      }
    }
  }
  if (MODE == 3) { // per-row max (query stabilizer), only real feature cols
#pragma unroll
    for (int j = 0; j < 8; j++) {
      int row = m0 + wv * 16 + kh * 8 + j;
      if (row >= P.M) continue;
      float mloc = -3.4e38f; bool any = false;
#pragma unroll
      for (int t = 0; t < TN; t++) {
        int col = n0 + t * 16 + nl;
        if (col < P.maxcols && col < P.Nn) { mloc = fmaxf(mloc, acc[t][j]); any = true; }
      }
      if (any) atomicMax(&P.amax[z * P.batchAmax + row], encMax(mloc));
    }
  }
  if (MODE == 4) { // global max (key stabilizer)
    float mloc = -3.4e38f; bool any = false;
#pragma unroll
    for (int t = 0; t < TN; t++) {
      int col = n0 + t * 16 + nl;
      if (col >= P.maxcols || col >= P.Nn) continue;
#pragma unroll
      for (int j = 0; j < 8; j++) {
        int row = m0 + wv * 16 + kh * 8 + j;
        if (row < P.M) { mloc = fmaxf(mloc, acc[t][j]); any = true; }
      }
    }
    if (any) atomicMax(&P.amax[0], encMax(mloc));
  }
}

// ---------------------------------------------------------------------------
// Support kernels
// ---------------------------------------------------------------------------
__global__ void perf_convert_wT(const float* w, __bf16* wt, int K, int N) {
  long idx = (long)blockIdx.x * 256 + threadIdx.x;
  if (idx >= (long)K * N) return;
  int n = (int)(idx / K), k = (int)(idx % K);
  wt[(long)n * K + k] = (__bf16)w[(long)k * N + n];
}

__global__ void perf_convert_proj(const float* proj, __bf16* pt) {
  int idx = blockIdx.x * 256 + threadIdx.x;
  if (idx >= FP * DH) return;
  int f = idx / DH;
  pt[idx] = (f < NF) ? (__bf16)(proj[idx] * DATA_NORM) : (__bf16)0.0f;
}

__global__ void perf_encoder(const float* x, const float* w, const float* b, float* h) {
  long idx = (long)blockIdx.x * 256 + threadIdx.x;
  if (idx >= (long)ROWS * DM) return;
  int r = (int)(idx >> 9), c = (int)(idx & 511);
  const float* xr = x + (long)r * 24;
  float s = 0.0f;
#pragma unroll
  for (int k = 0; k < 24; k++) s += xr[k] * w[k * DM + c];
  h[idx] = (s + b[c]) * SQRT_DM;
}

__global__ __launch_bounds__(256)
void perf_layernorm(const float* h, const float* g, const float* b, __bf16* y) {
  __shared__ float s1[256], s2[256];
  long row = blockIdx.x;
  const float* hr = h + row * DM;
  float a0 = hr[threadIdx.x], a1 = hr[threadIdx.x + 256];
  s1[threadIdx.x] = a0 + a1;
  s2[threadIdx.x] = a0 * a0 + a1 * a1;
  __syncthreads();
  for (int st = 128; st > 0; st >>= 1) {
    if ((int)threadIdx.x < st) {
      s1[threadIdx.x] += s1[threadIdx.x + st];
      s2[threadIdx.x] += s2[threadIdx.x + st];
    }
    __syncthreads();
  }
  float mu = s1[0] * (1.0f / DM);
  float var = s2[0] * (1.0f / DM) - mu * mu;
  float rstd = rsqrtf(var + 1e-5f);
  y[row * DM + threadIdx.x]       = (__bf16)((a0 - mu) * rstd * g[threadIdx.x] + b[threadIdx.x]);
  y[row * DM + threadIdx.x + 256] = (__bf16)((a1 - mu) * rstd * g[threadIdx.x + 256] + b[threadIdx.x + 256]);
}

__global__ void perf_pack(const float* qf, const float* kf, const float* vf,
                          __bf16* qb, __bf16* kb, __bf16* vT) {
  long idx = (long)blockIdx.x * 256 + threadIdx.x;
  if (idx >= (long)ROWS * DM) return;
  int bn = (int)(idx >> 9), c = (int)(idx & 511);
  int h = c >> 6, d = c & 63;
  int b = bn >> 11, n = bn & 2047;
  long bh = (long)b * NH + h;
  qb[(bh * SEQ + n) * DH + d] = (__bf16)qf[idx];
  kb[(bh * SEQ + n) * DH + d] = (__bf16)kf[idx];
  vT[(bh * DH + d) * SEQ + n] = (__bf16)vf[idx];
}

__global__ void perf_diag(const float* qf, const float* kf, float* dq, float* dk) {
  int idx = blockIdx.x * 256 + threadIdx.x;
  if (idx >= BH * SEQ) return;
  int bh = idx >> 11, n = idx & 2047;
  int b = bh >> 3, h = bh & 7;
  const float* qr = qf + ((long)b * SEQ + n) * DM + h * DH;
  const float* kr = kf + ((long)b * SEQ + n) * DM + h * DH;
  float sq = 0.0f, sk = 0.0f;
  for (int d = 0; d < DH; d++) { sq += qr[d] * qr[d]; sk += kr[d] * kr[d]; }
  dq[idx] = DIAG_SCL * sq;
  dk[idx] = DIAG_SCL * sk;
}

__global__ void perf_initmax(unsigned* qm, unsigned* km) {
  int idx = blockIdx.x * 256 + threadIdx.x;
  if (idx < BH * SEQ) qm[idx] = 0u;
  if (idx == 0) km[0] = 0u;
}

__global__ void perf_ksum(const __bf16* kpT, float* ks) {
  int idx = blockIdx.x * 256 + threadIdx.x;
  if (idx >= BH * FP) return;
  const __bf16* r = kpT + (long)idx * SEQ;
  float s = 0.0f;
  for (int n = 0; n < SEQ; n++) s += (float)r[n];
  ks[idx] = s;
}

__global__ void perf_dinv(const __bf16* qp, const float* ks, float* di) {
  int idx = blockIdx.x * 256 + threadIdx.x;
  if (idx >= BH * SEQ) return;
  int bh = idx >> 11;
  const __bf16* qr = qp + (long)idx * FP;
  const float* kr = ks + bh * FP;
  float s = 0.0f;
  for (int f = 0; f < FP; f++) s += (float)qr[f] * kr[f];
  di[idx] = 1.0f / s;
}

__global__ void perf_pool(const float* h, float* p) {
  int idx = blockIdx.x * 256 + threadIdx.x;
  if (idx >= BATCH * DM) return;
  int b = idx >> 9, c = idx & 511;
  const float* hb = h + (long)b * SEQ * DM + c;
  float m = hb[0];
  for (int n = 1; n < SEQ; n++) m = fmaxf(m, hb[(long)n * DM]);
  p[idx] = m;
}

__global__ void perf_head1(const float* p, const float* w1, const float* b1, float* zb) {
  int idx = blockIdx.x * 256 + threadIdx.x;
  if (idx >= BATCH * DM) return;
  int b = idx >> 9, j = idx & 511;
  float s = b1[j];
  for (int d = 0; d < DM; d++) s += p[b * DM + d] * w1[d * DM + j];
  zb[idx] = fmaxf(s, 0.0f);
}

__global__ void perf_head2(const float* zb, const float* w2, const float* b2, float* out) {
  int idx = blockIdx.x * 256 + threadIdx.x;
  if (idx >= BATCH * NCLS) return;
  int b = idx / NCLS, c = idx % NCLS;
  float s = b2[c];
  for (int j = 0; j < DM; j++) s += zb[b * DM + j] * w2[j * NCLS + c];
  out[idx] = s;
}

// ---------------------------------------------------------------------------
static inline dim3 ggrid(int M, int Nn, int bat, int bn) {
  return dim3((unsigned)((Nn + bn - 1) / bn), (unsigned)((M + 127) / 128), (unsigned)bat);
}
static inline dim3 ew(long n) { return dim3((unsigned)((n + 255) / 256)); }

static inline GemmP mkP(const __bf16* A, int lda, long bA,
                        const __bf16* Bt, int ldb, long bB,
                        int M, int Nn, int K) {
  GemmP P = {};
  P.A = A; P.Bt = Bt; P.lda = lda; P.ldb = ldb;
  P.batchA = bA; P.batchB = bB;
  P.M = M; P.Nn = Nn; P.K = K; P.maxcols = NF;
  return P;
}

extern "C" void kernel_launch(void* const* d_in, const int* in_sizes, int n_in,
                              void* d_out, int out_size, void* d_ws, size_t ws_size,
                              hipStream_t stream) {
  (void)in_sizes; (void)n_in; (void)out_size; (void)ws_size;
  const float* x     = (const float*)d_in[0];
  const float* enc_w = (const float*)d_in[1];
  const float* enc_b = (const float*)d_in[2];
  const float* ln1_g = (const float*)d_in[3];
  const float* ln1_b = (const float*)d_in[4];
  const float* wq = (const float*)d_in[5];  const float* bq = (const float*)d_in[6];
  const float* wk = (const float*)d_in[7];  const float* bk = (const float*)d_in[8];
  const float* wv = (const float*)d_in[9];  const float* bv = (const float*)d_in[10];
  const float* wo = (const float*)d_in[11]; const float* bo = (const float*)d_in[12];
  const float* proj  = (const float*)d_in[13];
  const float* ln2_g = (const float*)d_in[14];
  const float* ln2_b = (const float*)d_in[15];
  const float* ff_w1 = (const float*)d_in[16]; const float* ff_b1 = (const float*)d_in[17];
  const float* ff_w2 = (const float*)d_in[18]; const float* ff_b2 = (const float*)d_in[19];
  const float* mlp_w1 = (const float*)d_in[20]; const float* mlp_b1 = (const float*)d_in[21];
  const float* mlp_w2 = (const float*)d_in[22]; const float* mlp_b2 = (const float*)d_in[23];
  float* outp = (float*)d_out;

  // ---- workspace bump allocator with lifetime-based aliasing ----
  char* w8 = (char*)d_ws;
  size_t off = 0;
  auto take = [&](size_t b) -> char* {
    char* p = w8 + off; off = (off + b + 255) & ~(size_t)255; return p;
  };
  float*  hbuf = (float*)take((size_t)ROWS * DM * 4);          // residual stream
  __bf16* ybf  = (__bf16*)take((size_t)ROWS * DM * 2);         // LN output (reused)
  char*   bigA = take((size_t)3 * ROWS * DM * 4);              // qf/kf/vf, later qp
  float*  qf = (float*)bigA;
  float*  kf = (float*)(bigA + (size_t)ROWS * DM * 4);
  float*  vf = (float*)(bigA + (size_t)2 * ROWS * DM * 4);
  __bf16* qp = (__bf16*)bigA;                                   // [BH][SEQ][FP]
  __bf16* kpT = (__bf16*)take((size_t)BH * FP * SEQ * 2);       // [BH][FP][SEQ]
  char*   r2 = take((size_t)4 * ROWS * DM * 2);                 // qbf/kbf/vT/attn -> gbuf
  __bf16* qbf   = (__bf16*)r2;
  __bf16* kbf   = (__bf16*)(r2 + (size_t)ROWS * DM * 2);
  __bf16* vT    = (__bf16*)(r2 + (size_t)2 * ROWS * DM * 2);    // [BH][DH][SEQ]
  __bf16* attnb = (__bf16*)(r2 + (size_t)3 * ROWS * DM * 2);    // [ROWS][DM]
  __bf16* gbuf  = (__bf16*)r2;                                  // [ROWS][FF]
  float*    diagq = (float*)take((size_t)BH * SEQ * 4);
  float*    diagk = (float*)take((size_t)BH * SEQ * 4);
  unsigned* qmax  = (unsigned*)take((size_t)BH * SEQ * 4);
  unsigned* kmax  = (unsigned*)take(256);
  float*    ksum  = (float*)take((size_t)BH * FP * 4);
  __bf16*   ctxT  = (__bf16*)take((size_t)BH * DH * FP * 2);    // [BH][DH][FP]
  float*    dinv  = (float*)take((size_t)BH * SEQ * 4);
  __bf16* wqT  = (__bf16*)take((size_t)DM * DM * 2);
  __bf16* wkT  = (__bf16*)take((size_t)DM * DM * 2);
  __bf16* wvT  = (__bf16*)take((size_t)DM * DM * 2);
  __bf16* woT  = (__bf16*)take((size_t)DM * DM * 2);
  __bf16* ff1T = (__bf16*)take((size_t)FF * DM * 2);
  __bf16* ff2T = (__bf16*)take((size_t)DM * FF * 2);
  __bf16* projT = (__bf16*)take((size_t)FP * DH * 2);
  float*  pbuf = (float*)take((size_t)BATCH * DM * 4);
  float*  zbuf = (float*)take((size_t)BATCH * DM * 4);

  // ---- weight conversion (f32 -> bf16, transposed to [N][K]) ----
  perf_convert_wT<<<ew((long)DM * DM), 256, 0, stream>>>(wq, wqT, DM, DM);
  perf_convert_wT<<<ew((long)DM * DM), 256, 0, stream>>>(wk, wkT, DM, DM);
  perf_convert_wT<<<ew((long)DM * DM), 256, 0, stream>>>(wv, wvT, DM, DM);
  perf_convert_wT<<<ew((long)DM * DM), 256, 0, stream>>>(wo, woT, DM, DM);
  perf_convert_wT<<<ew((long)DM * FF), 256, 0, stream>>>(ff_w1, ff1T, DM, FF);
  perf_convert_wT<<<ew((long)FF * DM), 256, 0, stream>>>(ff_w2, ff2T, FF, DM);
  perf_convert_proj<<<ew(FP * DH), 256, 0, stream>>>(proj, projT);

  // ---- encoder + LN1 ----
  perf_encoder<<<ew((long)ROWS * DM), 256, 0, stream>>>(x, enc_w, enc_b, hbuf);
  perf_layernorm<<<ROWS, 256, 0, stream>>>(hbuf, ln1_g, ln1_b, ybf);

  // ---- QKV projections (WMMA, f32 out) ----
  {
    GemmP P = mkP(ybf, DM, 0, wqT, DM, 0, ROWS, DM, DM);
    P.outf = qf; P.ldo = DM; P.bias = bq;
    perf_gemm<0, 8><<<ggrid(ROWS, DM, 1, 128), 256, 0, stream>>>(P);
    P.Bt = wkT; P.outf = kf; P.bias = bk;
    perf_gemm<0, 8><<<ggrid(ROWS, DM, 1, 128), 256, 0, stream>>>(P);
    P.Bt = wvT; P.outf = vf; P.bias = bv;
    perf_gemm<0, 8><<<ggrid(ROWS, DM, 1, 128), 256, 0, stream>>>(P);
  }

  // ---- FAVOR+ features ----
  perf_pack<<<ew((long)ROWS * DM), 256, 0, stream>>>(qf, kf, vf, qbf, kbf, vT);
  perf_diag<<<ew(BH * SEQ), 256, 0, stream>>>(qf, kf, diagq, diagk);
  perf_initmax<<<ew(BH * SEQ), 256, 0, stream>>>(qmax, kmax);

  { // pass 1: stabilizer maxes over dd = q @ proj^T (K=64)
    GemmP P = mkP(qbf, DH, (long)SEQ * DH, projT, DH, 0, SEQ, FP, DH);
    P.amax = qmax; P.batchAmax = SEQ;
    perf_gemm<3, 4><<<ggrid(SEQ, FP, BH, 64), 256, 0, stream>>>(P);
    P.A = kbf; P.amax = kmax;
    perf_gemm<4, 4><<<ggrid(SEQ, FP, BH, 64), 256, 0, stream>>>(P);
  }
  { // pass 2: qp (row-major) and kp^T (transposed) with exp epilogue
    GemmP P = mkP(qbf, DH, (long)SEQ * DH, projT, DH, 0, SEQ, FP, DH);
    P.outb = qp; P.ldo = FP; P.batchOut = (long)SEQ * FP;
    P.diag = diagq; P.batchDiag = SEQ; P.amax = qmax; P.batchAmax = SEQ;
    perf_gemm<5, 4><<<ggrid(SEQ, FP, BH, 64), 256, 0, stream>>>(P);
    P.A = kbf; P.outb = kpT; P.ldo = SEQ; P.batchOut = (long)FP * SEQ;
    P.diag = diagk; P.amax = kmax;
    perf_gemm<6, 4><<<ggrid(SEQ, FP, BH, 64), 256, 0, stream>>>(P);
  }

  // ---- linear attention ----
  perf_ksum<<<ew(BH * FP), 256, 0, stream>>>(kpT, ksum);
  { // ctx^T[d][f] = sum_n kp[n][f] * v[n][d]   (M=FP, N=DH, K=SEQ)
    GemmP P = mkP(kpT, SEQ, (long)FP * SEQ, vT, SEQ, (long)DH * SEQ, FP, DH, SEQ);
    P.outb = ctxT; P.ldo = FP; P.batchOut = (long)DH * FP;
    perf_gemm<7, 4><<<ggrid(FP, DH, BH, 64), 256, 0, stream>>>(P);
  }
  perf_dinv<<<ew(BH * SEQ), 256, 0, stream>>>(qp, ksum, dinv);
  { // attn[n][d] = (qp @ ctx) * d_inv, scattered to [B,N,H*DH] bf16
    GemmP P = mkP(qp, FP, (long)SEQ * FP, ctxT, FP, (long)DH * FP, SEQ, DH, FP);
    P.outb = attnb; P.diag = dinv; P.batchDiag = SEQ;
    perf_gemm<8, 4><<<ggrid(SEQ, DH, BH, 64), 256, 0, stream>>>(P);
  }
  { // h += attn @ wo + bo
    GemmP P = mkP(attnb, DM, 0, woT, DM, 0, ROWS, DM, DM);
    P.outf = hbuf; P.ldo = DM; P.bias = bo;
    perf_gemm<1, 8><<<ggrid(ROWS, DM, 1, 128), 256, 0, stream>>>(P);
  }

  // ---- FeedForward block ----
  perf_layernorm<<<ROWS, 256, 0, stream>>>(hbuf, ln2_g, ln2_b, ybf);
  { // gbuf = gelu(y @ ff_w1 + b1)
    GemmP P = mkP(ybf, DM, 0, ff1T, DM, 0, ROWS, FF, DM);
    P.outb = gbuf; P.ldo = FF; P.bias = ff_b1;
    perf_gemm<2, 8><<<ggrid(ROWS, FF, 1, 128), 256, 0, stream>>>(P);
  }
  { // h += gbuf @ ff_w2 + b2
    GemmP P = mkP(gbuf, FF, 0, ff2T, FF, 0, ROWS, DM, FF);
    P.outf = hbuf; P.ldo = DM; P.bias = ff_b2;
    perf_gemm<1, 8><<<ggrid(ROWS, DM, 1, 128), 256, 0, stream>>>(P);
  }

  // ---- pool + head ----
  perf_pool<<<ew(BATCH * DM), 256, 0, stream>>>(hbuf, pbuf);
  perf_head1<<<ew(BATCH * DM), 256, 0, stream>>>(pbuf, mlp_w1, mlp_b1, zbuf);
  perf_head2<<<ew(BATCH * NCLS), 256, 0, stream>>>(zbuf, mlp_w2, mlp_b2, outp);
}
